// Ernie4_5_MoeSparseMoeBlock_53901839565699
// MI455X (gfx1250) — matrix operand order repaired and checked by reading, not compiled, not run on hardware
//
#include <hip/hip_runtime.h>
#include <math.h>

#define TOK   2048
#define HDIM  1024
#define NEXP  16
#define TOPK  4
#define IDIM  512   // unified intermediate tile; shared expert = 2 halves of 512

typedef __attribute__((ext_vector_type(16))) __bf16        v16bf;
typedef __attribute__((ext_vector_type(2)))  __bf16        v2bf;
typedef __attribute__((ext_vector_type(8)))  float         v8f;
typedef __attribute__((ext_vector_type(2)))  float         v2f;
typedef __attribute__((ext_vector_type(4)))  unsigned int  u32x4;
typedef __attribute__((ext_vector_type(8)))  int           i32x8;
typedef __attribute__((ext_vector_type(4)))  int           i32x4;

#define HAVE_TDM __has_builtin(__builtin_amdgcn_tensor_load_to_lds)

union Frag16 { v16bf v; uint4 q[2]; };

__device__ inline unsigned int pack_bf16(float a, float b) {
    v2f f = {a, b};
    v2bf h = __builtin_convertvector(f, v2bf);   // -> v_cvt_pk_bf16_f32
    return __builtin_bit_cast(unsigned int, h);
}
__device__ inline unsigned short to_bf16(float a) {
    return __builtin_bit_cast(unsigned short, (__bf16)a);
}

// ---------------------------------------------------------------------------
// TDM: DMA a [64 rows x 32 bf16] panel (row stride LD elements, compile-time)
// from global into LDS, padding each 64B row with 16B -> 80B LDS row stride
// (16B-aligned rows for ds_load_b128). Tracked by TENSORcnt. With LD a
// template constant, descriptor group 1 is a pure build-time constant.
// ---------------------------------------------------------------------------
template <int LD>
__device__ inline void tdm_load_panel(unsigned lds_off, const unsigned short* gptr) {
#if HAVE_TDM
    unsigned long long ga = (unsigned long long)(const void*)gptr;
    constexpr unsigned td0 = (unsigned)LD;
    constexpr unsigned td1 = 0x7FFFFFFFu;
    u32x4 g0;
    g0[0] = 1u;                                                 // count=1
    g0[1] = lds_off;                                            // lds_addr
    g0[2] = (unsigned)(ga & 0xFFFFFFFFull);
    g0[3] = (unsigned)((ga >> 32) & 0x1FFFFFFull) | (2u << 30); // hi | type=2
    // data_size=2B | pad_enable | pad_interval=16 DWORDs | pad_amount=4 DWORDs
    constexpr i32x8 g1 = {
        (int)((1u << 16) | (1u << 20) | (3u << 22) | (3u << 25)),
        (int)((td0 & 0xFFFFu) << 16),                           // tensor_dim0 lo
        (int)((td0 >> 16) | ((td1 & 0xFFFFu) << 16)),           // dim0 hi|dim1 lo
        (int)((td1 >> 16) | (32u << 16)),                       // dim1 hi|tile_dim0
        64,                                                     // tile_dim1
        (int)td0,                                               // dim0_stride lo
        0, 0};
    i32x4 z4 = {0, 0, 0, 0};
#if defined(__clang_major__) && __clang_major__ >= 23
    i32x8 z8 = {0, 0, 0, 0, 0, 0, 0, 0};
    __builtin_amdgcn_tensor_load_to_lds(g0, g1, z4, z4, z8, 0);
#else
    __builtin_amdgcn_tensor_load_to_lds(g0, g1, z4, z4, 0);
#endif
#else
    (void)lds_off; (void)gptr;
#endif
}

#if !HAVE_TDM
__device__ inline void stage_panel(unsigned short* dst, const unsigned short* g,
                                   int ld, int tid) {
#pragma unroll
    for (int i = 0; i < 2; ++i) {
        int idx = tid + i * 256;
        int row = idx >> 3, c4 = (idx & 7) * 4;
        *(uint2*)&dst[row * 40 + c4] = *(const uint2*)&g[(size_t)row * ld + c4];
    }
}
#endif

// ---------------------------------------------------------------------------
__global__ void zero_kernel(float* __restrict__ out, int n, int* __restrict__ counts) {
    int i = blockIdx.x * blockDim.x + threadIdx.x;
    for (int k = i; k < n; k += gridDim.x * blockDim.x) out[k] = 0.0f;
    if (i < NEXP) counts[i] = 0;
}

// ---------------------------------------------------------------------------
// Transpose + fp32->bf16 convert: src [R,C] fp32 -> dst [C,R] bf16 (batched).
// ---------------------------------------------------------------------------
__global__ __launch_bounds__(256) void transpose_cvt(const float* __restrict__ src,
                                                     unsigned short* __restrict__ dst,
                                                     int R, int C) {
    __shared__ float t[32][33];
    size_t bo = (size_t)blockIdx.z * R * C;
    src += bo; dst += bo;
    const int r0 = blockIdx.y * 32, c0 = blockIdx.x * 32;
    const int tid = threadIdx.x;
    {
        int r = tid >> 3, c4 = (tid & 7) * 4;
        const float4 f = *(const float4*)&src[(size_t)(r0 + r) * C + c0 + c4];
        t[r][c4] = f.x; t[r][c4 + 1] = f.y; t[r][c4 + 2] = f.z; t[r][c4 + 3] = f.w;
    }
    __syncthreads();
    {
        int cc = tid >> 3, rr = (tid & 7) * 4;
        uint2 p;
        p.x = pack_bf16(t[rr][cc], t[rr + 1][cc]);
        p.y = pack_bf16(t[rr + 2][cc], t[rr + 3][cc]);
        *(uint2*)&dst[(size_t)(c0 + cc) * R + r0 + rr] = p;
    }
}

// ---------------------------------------------------------------------------
// Router: one wave per token.
// ---------------------------------------------------------------------------
__global__ void router_kernel(const float* __restrict__ x,
                              const float* __restrict__ gw,
                              const float* __restrict__ bias,
                              float* __restrict__ logits_out,
                              float* __restrict__ cw,
                              int*   __restrict__ counts,
                              int*   __restrict__ lists) {
    int wave = threadIdx.x >> 5;
    int lane = threadIdx.x & 31;
    int t = blockIdx.x * (blockDim.x >> 5) + wave;
    if (t >= TOK) return;

    float acc[NEXP];
#pragma unroll
    for (int e = 0; e < NEXP; ++e) acc[e] = 0.0f;
    for (int h = lane; h < HDIM; h += 32) {
        float xv = x[(size_t)t * HDIM + h];
        const float* g = gw + (size_t)h * NEXP;
#pragma unroll
        for (int e = 0; e < NEXP; ++e) acc[e] = fmaf(xv, g[e], acc[e]);
    }
#pragma unroll
    for (int e = 0; e < NEXP; ++e)
        for (int off = 16; off > 0; off >>= 1) acc[e] += __shfl_xor(acc[e], off, 32);

    if (lane == 0) {
        float m = acc[0];
        for (int e = 1; e < NEXP; ++e) m = fmaxf(m, acc[e]);
        float p[NEXP]; float s = 0.0f;
        for (int e = 0; e < NEXP; ++e) { p[e] = expf(acc[e] - m); s += p[e]; }
        float inv = 1.0f / s;
        for (int e = 0; e < NEXP; ++e) p[e] *= inv;
        for (int e = 0; e < NEXP; ++e) logits_out[(size_t)t * NEXP + e] = acc[e];

        float sc[NEXP];
        for (int e = 0; e < NEXP; ++e) sc[e] = p[e] + bias[e];
        int sel[TOPK]; float rw[TOPK]; float rs = 0.0f;
        for (int k = 0; k < TOPK; ++k) {
            int best = 0; float bv = -1e30f;
            for (int e = 0; e < NEXP; ++e)
                if (sc[e] > bv) { bv = sc[e]; best = e; }
            sel[k] = best; rw[k] = p[best]; rs += p[best]; sc[best] = -1e30f;
        }
        float rinv = 1.0f / fmaxf(rs, 1e-12f);
        float cwl[NEXP];
        for (int e = 0; e < NEXP; ++e) cwl[e] = 0.0f;
        for (int k = 0; k < TOPK; ++k) {
            int e = sel[k];
            cwl[e] += rw[k] * rinv;
            int slot = atomicAdd(&counts[e], 1);
            lists[e * TOK + slot] = t;
        }
        for (int e = 0; e < NEXP; ++e) cw[(size_t)t * NEXP + e] = cwl[e];
    }
}

// ---------------------------------------------------------------------------
// Fused gated MLP over a 32-token tile. 8 waves = 2 row-groups x 4 col-groups.
// Weights pre-transposed bf16 [N,K]; panels DMA'd by TDM (double-buffered);
// fragments are raw ds_load_b128; WMMA f32_16x16x32_bf16 accumulation.
// ---------------------------------------------------------------------------
template <bool GATHER, int LDGU, int LDD>
__global__ __launch_bounds__(256) void mlp_kernel(
    const float* __restrict__ x,
    const unsigned short* __restrict__ WgT,  // [N=I][K=H] bf16, + y*gStride
    const unsigned short* __restrict__ WuT,
    const unsigned short* __restrict__ WdT,  // [N=H][K=I] bf16, + y*dStride
    long gStride, long dStride,
    const float* __restrict__ cw, const int* __restrict__ counts,
    const int* __restrict__ lists, float* __restrict__ out) {
    constexpr int MT = 32, NP = 64, AS = 40, AI = IDIM + 8, BS = 40;
    constexpr int KS1 = HDIM / 32, NCB1 = IDIM / NP, S1 = KS1 * NCB1;
    constexpr int KS2 = IDIM / 32, NCB2 = HDIM / NP, S2 = KS2 * NCB2;
    constexpr unsigned PB = 64 * BS * 2;  // panel bytes

    __shared__ __align__(16) unsigned short sA[MT * AS];
    __shared__ __align__(16) unsigned short sAct[MT * AI];
    __shared__ __align__(16) unsigned short sBG[2][64 * BS];
    __shared__ __align__(16) unsigned short sBU[2][64 * BS];
    __shared__ int   sTok[MT];
    __shared__ float sW[MT];

    const int y    = blockIdx.y;
    const int cnt  = GATHER ? counts[y] : TOK;
    const int tile = blockIdx.x;
    if (tile * MT >= cnt) return;  // uniform: EXEC all-ones for WMMA

    const unsigned short* Wg = WgT + (size_t)y * gStride;
    const unsigned short* Wu = WuT + (size_t)y * gStride;
    const unsigned short* Wd = WdT + (size_t)y * dStride;

    const int tid  = threadIdx.x;
    const int wid  = tid >> 5;
    const int lane = tid & 31;
    const int half = lane >> 4;
    const int ln   = lane & 15;
    const int rg   = wid >> 2;
    const int cg   = wid & 3;

    const unsigned ldsBG0 = (unsigned)(unsigned long long)(size_t)(void*)&sBG[0][0];
    const unsigned ldsBU0 = (unsigned)(unsigned long long)(size_t)(void*)&sBU[0][0];

    if (tid < MT) {
        int m = tile * MT + tid;
        int tk;
        if (GATHER) tk = lists[y * TOK + ((m < cnt) ? m : (tile * MT))];
        else        tk = m;
        sTok[tid] = tk;
        sW[tid] = GATHER ? ((m < cnt) ? cw[(size_t)tk * NEXP + y] : 0.0f) : 1.0f;
    }
    __syncthreads();

    // Hoisted A-staging addresses (loop-invariant gather).
    const float* xA; int aoff;
    {
        int row = tid >> 3, kk = (tid & 7) * 4;
        xA = x + (size_t)sTok[row] * HDIM + kk;
        aoff = row * AS + kk;
    }

    // ---- Phase 1: act = silu(x@Wg) * (x@Wu) * weight ----------------------
    v8f accG[NCB1] = {};
    v8f accU[NCB1] = {};
#if HAVE_TDM
    if (wid == 0) {
        tdm_load_panel<LDGU>(ldsBG0, Wg);
        tdm_load_panel<LDGU>(ldsBU0, Wu);
    }
#endif
    Frag16 fa;
    int s = 0;
    for (int ks = 0; ks < KS1; ++ks) {
        {  // stage A: 32 rows x 32 K, fp32 -> bf16 (one cvt_pk pair per thread)
            const float4 f = *(const float4*)(xA + ks * 32);
            *(uint2*)&sA[aoff] = make_uint2(pack_bf16(f.x, f.y), pack_bf16(f.z, f.w));
        }
#pragma unroll
        for (int cb = 0; cb < NCB1; ++cb, ++s) {
            const int cur = s & 1;
#if HAVE_TDM
            if (wid == 0) {
                if (s + 1 < S1) {
                    int ns = s + 1;
                    int ncb = ns & (NCB1 - 1), nk = (ns / NCB1) * 32;
                    tdm_load_panel<LDGU>(ldsBG0 + (unsigned)(cur ^ 1) * PB,
                                         Wg + (size_t)(ncb * NP) * LDGU + nk);
                    tdm_load_panel<LDGU>(ldsBU0 + (unsigned)(cur ^ 1) * PB,
                                         Wu + (size_t)(ncb * NP) * LDGU + nk);
                    __builtin_amdgcn_s_wait_tensorcnt(2);
                } else {
                    __builtin_amdgcn_s_wait_tensorcnt(0);
                }
            }
#else
            stage_panel(sBG[cur], Wg + (size_t)(cb * NP) * LDGU + ks * 32, LDGU, tid);
            stage_panel(sBU[cur], Wu + (size_t)(cb * NP) * LDGU + ks * 32, LDGU, tid);
#endif
            __syncthreads();
            if (cb == 0) {
                fa.q[0] = *(const uint4*)&sA[(rg * 16 + ln) * AS + half * 16];
                fa.q[1] = *(const uint4*)&sA[(rg * 16 + ln) * AS + half * 16 + 8];
            }
            Frag16 fg, fu;
            const unsigned short* bg = &sBG[cur][(cg * 16 + ln) * BS + half * 16];
            fg.q[0] = *(const uint4*)bg;
            fg.q[1] = *(const uint4*)(bg + 8);
            const unsigned short* bu = &sBU[cur][(cg * 16 + ln) * BS + half * 16];
            fu.q[0] = *(const uint4*)bu;
            fu.q[1] = *(const uint4*)(bu + 8);
            accG[cb] = __builtin_amdgcn_wmma_f32_16x16x32_bf16(
                false, fa.v, false, fg.v, (short)0, accG[cb], false, false);
            accU[cb] = __builtin_amdgcn_wmma_f32_16x16x32_bf16(
                false, fa.v, false, fu.v, (short)0, accU[cb], false, false);
            __syncthreads();
        }
    }
#pragma unroll
    for (int cb = 0; cb < NCB1; ++cb) {
        int col = cb * NP + cg * 16 + ln;
#pragma unroll
        for (int j = 0; j < 8; ++j) {
            int m = rg * 16 + half * 8 + j;
            float g = accG[cb][j], u = accU[cb][j];
            float a = g * __builtin_amdgcn_rcpf(1.0f + __expf(-g)) * u * sW[m];
            sAct[m * AI + col] = to_bf16(a);
        }
    }
    __syncthreads();

    // ---- Phase 2: out += act @ Wd -----------------------------------------
    v8f acc2[NCB2] = {};
#if HAVE_TDM
    if (wid == 0) tdm_load_panel<LDD>(ldsBG0, Wd);
#endif
    int s2 = 0;
    for (int ks = 0; ks < KS2; ++ks) {
#pragma unroll
        for (int cb = 0; cb < NCB2; ++cb, ++s2) {
            const int cur = s2 & 1;
#if HAVE_TDM
            if (wid == 0) {
                if (s2 + 1 < S2) {
                    int ns = s2 + 1;
                    int ncb = ns & (NCB2 - 1), nk = (ns / NCB2) * 32;
                    tdm_load_panel<LDD>(ldsBG0 + (unsigned)(cur ^ 1) * PB,
                                        Wd + (size_t)(ncb * NP) * LDD + nk);
                    __builtin_amdgcn_s_wait_tensorcnt(1);
                } else {
                    __builtin_amdgcn_s_wait_tensorcnt(0);
                }
            }
#else
            stage_panel(sBG[cur], Wd + (size_t)(cb * NP) * LDD + ks * 32, LDD, tid);
#endif
            __syncthreads();
            if (cb == 0) {
                fa.q[0] = *(const uint4*)&sAct[(rg * 16 + ln) * AI + ks * 32 + half * 16];
                fa.q[1] = *(const uint4*)&sAct[(rg * 16 + ln) * AI + ks * 32 + half * 16 + 8];
            }
            Frag16 fb;
            const unsigned short* bd = &sBG[cur][(cg * 16 + ln) * BS + half * 16];
            fb.q[0] = *(const uint4*)bd;
            fb.q[1] = *(const uint4*)(bd + 8);
            acc2[cb] = __builtin_amdgcn_wmma_f32_16x16x32_bf16(
                false, fa.v, false, fb.v, (short)0, acc2[cb], false, false);
            __syncthreads();
        }
    }
#pragma unroll
    for (int cb = 0; cb < NCB2; ++cb) {
        int col = cb * NP + cg * 16 + ln;
#pragma unroll
        for (int j = 0; j < 8; ++j) {
            int m = rg * 16 + half * 8 + j;
            atomicAdd(&out[(size_t)sTok[m] * HDIM + col], acc2[cb][j]);
        }
    }
}

// ---------------------------------------------------------------------------
extern "C" void kernel_launch(void* const* d_in, const int* in_sizes, int n_in,
                              void* d_out, int out_size, void* d_ws, size_t ws_size,
                              hipStream_t stream) {
    (void)in_sizes; (void)n_in; (void)out_size; (void)ws_size;

    const float* x      = (const float*)d_in[0];
    const float* gw     = (const float*)d_in[1];
    const float* cbias  = (const float*)d_in[2];
    const float* gate_w = (const float*)d_in[3];
    const float* up_w   = (const float*)d_in[4];
    const float* down_w = (const float*)d_in[5];
    const float* sgw    = (const float*)d_in[6];
    const float* suw    = (const float*)d_in[7];
    const float* sdw    = (const float*)d_in[8];

    float* out_final  = (float*)d_out;
    float* out_logits = out_final + (size_t)TOK * HDIM;

    char* ws = (char*)d_ws;
    size_t o = 0;
    float* cw     = (float*)(ws + o); o += (size_t)TOK * NEXP * 4;
    int*   counts = (int*)(ws + o);   o += 256;
    int*   lists  = (int*)(ws + o);   o += (size_t)NEXP * TOK * 4;
    unsigned short* gT  = (unsigned short*)(ws + o); o += (size_t)NEXP * IDIM * HDIM * 2;
    unsigned short* uT  = (unsigned short*)(ws + o); o += (size_t)NEXP * IDIM * HDIM * 2;
    unsigned short* dT  = (unsigned short*)(ws + o); o += (size_t)NEXP * IDIM * HDIM * 2;
    unsigned short* sgT = (unsigned short*)(ws + o); o += (size_t)1024 * HDIM * 2;
    unsigned short* suT = (unsigned short*)(ws + o); o += (size_t)1024 * HDIM * 2;
    unsigned short* sdT = (unsigned short*)(ws + o); o += (size_t)1024 * HDIM * 2;

    zero_kernel<<<2048, 256, 0, stream>>>(out_final, TOK * HDIM, counts);
    router_kernel<<<TOK / 4, 128, 0, stream>>>(x, gw, cbias, out_logits, cw,
                                               counts, lists);

    // Pre-transpose + convert all weights to bf16 [N,K] layout.
    transpose_cvt<<<dim3(IDIM / 32, HDIM / 32, NEXP), 256, 0, stream>>>(gate_w, gT, HDIM, IDIM);
    transpose_cvt<<<dim3(IDIM / 32, HDIM / 32, NEXP), 256, 0, stream>>>(up_w,   uT, HDIM, IDIM);
    transpose_cvt<<<dim3(HDIM / 32, IDIM / 32, NEXP), 256, 0, stream>>>(down_w, dT, IDIM, HDIM);
    transpose_cvt<<<dim3(1024 / 32, HDIM / 32, 1),    256, 0, stream>>>(sgw, sgT, HDIM, 1024);
    transpose_cvt<<<dim3(1024 / 32, HDIM / 32, 1),    256, 0, stream>>>(suw, suT, HDIM, 1024);
    transpose_cvt<<<dim3(HDIM / 32, 1024 / 32, 1),    256, 0, stream>>>(sdw, sdT, 1024, HDIM);

    // Shared experts: SHARED_I=1024 as two I=512 halves.
    //   gate/up^T [1024][H]: half y -> row offset y*512 (stride 512*H), ld=H
    //   down^T    [H][1024]: half y -> col offset y*512, ld=1024
    mlp_kernel<false, HDIM, 1024><<<dim3(TOK / 32, 2), 256, 0, stream>>>(
        x, sgT, suT, sdT, 512L * HDIM, 512L,
        nullptr, nullptr, nullptr, out_final);
    // Routed experts: gate/up^T [E][I][H] (ld=H), down^T [E][H][I] (ld=I).
    mlp_kernel<true, HDIM, IDIM><<<dim3(TOK / 32, NEXP), 256, 0, stream>>>(
        x, gT, uT, dT, (long)IDIM * HDIM, (long)HDIM * IDIM,
        cw, counts, lists, out_final);
}
